// Router_49890340110394
// MI455X (gfx1250) — matrix operand order repaired and checked by reading, not compiled, not run on hardware
//
#include <hip/hip_runtime.h>
#include <hip/hip_bf16.h>

// ---------------------------------------------------------------------------
// MoE router for MI455X (gfx1250, wave32):
//   logits = x[8192,4096] @ W[4096,64];  top-8 per token; softmax over top-8.
// bf16x3 compensated WMMA GEMM (v_wmma_f32_16x16x32_bf16), B slabs staged into
// LDS by the Tensor Data Mover (double-buffered, padded rows via D# pad fields),
// fused top-k + softmax epilogue.
// ---------------------------------------------------------------------------

typedef __bf16 bf16_t;
typedef __attribute__((ext_vector_type(16))) __bf16 v16bf;
typedef __attribute__((ext_vector_type(8)))  __bf16 v8bf;
typedef __attribute__((ext_vector_type(8)))  float  v8f;
typedef __attribute__((ext_vector_type(4)))  unsigned int u32x4;
typedef __attribute__((ext_vector_type(8)))  int i32x8;
typedef __attribute__((ext_vector_type(4)))  int i32x4;

#define T_TOK  8192
#define D_DIM  4096
#define E_EXP  64
#define TOPK_K 8

#define WAVES_PB 4                    // waves per block
#define TOK_PW   16                   // tokens (M rows) per wave = one M tile
#define TOK_PB   (WAVES_PB * TOK_PW)  // 64 tokens per block -> 128 blocks
#define KPHASE   128                  // K span staged into LDS per phase
#define NPHASE   (D_DIM / KPHASE)     // 32 phases
#define KCHUNK   32                   // K per wmma_f32_16x16x32_bf16
#define LDS_ROW  (KPHASE + 8)         // +16B pad per 256B row -> conflict-free ds_load_b128
#define STAGE_BYTES (2 * E_EXP * LDS_ROW * 2)  // 2 planes * 64 cols * 272B = 34816

#if defined(__has_builtin)
#if __has_builtin(__builtin_amdgcn_tensor_load_to_lds) && \
    __has_builtin(__builtin_amdgcn_s_wait_tensorcnt)
#define USE_TDM 1
#endif
#endif

// ---------------------------------------------------------------------------
// top-8 (sorted desc, lax.top_k tie-breaking) + softmax, one thread per token.
// ---------------------------------------------------------------------------
__device__ __forceinline__ void topk8_softmax_store(const float* lg, int token,
                                                    float* __restrict__ out_w,
                                                    int* __restrict__ out_i) {
  float w[TOPK_K];
  int   id[TOPK_K];
#pragma unroll
  for (int k = 0; k < TOPK_K; ++k) { w[k] = -__builtin_inff(); id[k] = 0; }
  for (int e = 0; e < E_EXP; ++e) {
    float v = lg[e];
    if (v > w[TOPK_K - 1]) {            // strict > : earlier (lower) index wins ties
      w[TOPK_K - 1] = v; id[TOPK_K - 1] = e;
#pragma unroll
      for (int k = TOPK_K - 1; k > 0; --k) {
        if (w[k] > w[k - 1]) {
          float tw = w[k]; w[k] = w[k - 1]; w[k - 1] = tw;
          int   ti = id[k]; id[k] = id[k - 1]; id[k - 1] = ti;
        }
      }
    }
  }
  float m = w[0], s = 0.0f, ex[TOPK_K];
#pragma unroll
  for (int k = 0; k < TOPK_K; ++k) { ex[k] = __expf(w[k] - m); s += ex[k]; }
  float inv = 1.0f / s;
#pragma unroll
  for (int k = 0; k < TOPK_K; ++k) {
    out_w[token * TOPK_K + k] = ex[k] * inv;
    out_i[token * TOPK_K + k] = id[k];
  }
}

// ---------------------------------------------------------------------------
// Prep: W[D,E] f32 -> transposed bf16 planes bt[2][E][D] (hi, lo residual).
// ---------------------------------------------------------------------------
__global__ __launch_bounds__(256) void router_prep_b(const float* __restrict__ wde,
                                                     bf16_t* __restrict__ bt) {
  int idx = blockIdx.x * blockDim.x + threadIdx.x;   // idx = e*D + d
  if (idx >= E_EXP * D_DIM) return;
  int e = idx >> 12;          // / 4096
  int d = idx & (D_DIM - 1);
  float f  = wde[(size_t)d * E_EXP + e];
  __bf16 h = (__bf16)f;                      // RNE
  bt[idx]                         = h;       // hi plane
  bt[(size_t)E_EXP * D_DIM + idx] = (__bf16)(f - (float)h);  // lo plane
}

#if defined(USE_TDM)
// ---------------------------------------------------------------------------
// TDM descriptor (ISA 8.3-8.6): 2D tile, 128 rows x 256B, row stride 8KB,
// data_size=2B. pad_enable with pad_interval=5 (every 64 dwords = 256B) and
// pad_amount=3 (4 dwords = 16B) reproduces the LDS_ROW=136 padded layout.
// Issued by one wave; EXEC ignored; completion tracked with TENSORcnt.
// ---------------------------------------------------------------------------
__device__ __forceinline__ void tdm_stage_b(const bf16_t* gsrc, unsigned lds_byte) {
  unsigned long long ga = (unsigned long long)(uintptr_t)gsrc;
  u32x4 g0;
  g0[0] = 1u;                                  // count=1, user load descriptor
  g0[1] = lds_byte;                            // D#.lds_addr (bytes)
  g0[2] = (unsigned)(ga & 0xFFFFFFFFu);        // D#.global_addr[31:0]
  g0[3] = (unsigned)((ga >> 32) & 0x1FFFFFFu)  // D#.global_addr[56:32]
        | (2u << 30);                          // D#.type = 2 ("image")
  i32x8 g1;
  g1[0] = (1 << 16)                            // data_size = 1 -> 2-byte elements
        | (1 << 20)                            // pad_enable
        | (5 << 22)                            // pad_interval: 64 dwords (256B)
        | (3 << 25);                           // pad_amount: 4 dwords (16B)
  g1[1] = (int)(D_DIM << 16);                  // tensor_dim0[15:0] = 4096
  g1[2] = (int)((2 * E_EXP) << 16);            // tensor_dim0 hi=0 | tensor_dim1 = 128
  g1[3] = (int)(KPHASE << 16);                 // tensor_dim1 hi=0 | tile_dim0 = 128
  g1[4] = (int)(2 * E_EXP);                    // tile_dim1 = 128 | tile_dim2 = 0
  g1[5] = (int)D_DIM;                          // tensor_dim0_stride[31:0] = 4096 elems
  g1[6] = 0;                                   // stride0 hi | tensor_dim1_stride[15:0]
  g1[7] = (int)(((long long)D_DIM * 2 * E_EXP) >> 16);  // tensor_dim1_stride[47:16]
  i32x4 z4 = {0, 0, 0, 0};                     // groups 2/3 unused (2D tile)
  i32x8 z8 = {0, 0, 0, 0, 0, 0, 0, 0};         // extra operand (6-arg form): zero
  __builtin_amdgcn_tensor_load_to_lds(g0, g1, z4, z4, z8, 0);
}
#endif

// ---------------------------------------------------------------------------
// Main fused kernel: GEMM (bf16x3 WMMA) + top-k + softmax.
// ---------------------------------------------------------------------------
__global__ __launch_bounds__(WAVES_PB * 32) void router_gemm_topk(
    const float* __restrict__ x, const bf16_t* __restrict__ bt,
    float* __restrict__ out_w, int* __restrict__ out_i) {
  __shared__ __align__(16) unsigned char smem_raw[2 * STAGE_BYTES];  // double buffer
  float* sL = (float*)smem_raw;    // reused in epilogue: [token][68] logits

  const int tid  = threadIdx.x;
  const int lane = tid & 31;
  const int wave = tid >> 5;
  const int l15  = lane & 15;
  const int h16  = lane >> 4;                     // 0 for lanes 0-15, 1 for 16-31

  const int row = blockIdx.x * TOK_PB + wave * TOK_PW + l15;   // this lane's A row
  const float* xrow = x + (size_t)row * D_DIM;
  const int kloA = h16 ? 8 : 0;                   // A 16-bit layout half offset
  const int kloB = h16 ? 8 : 0;                   // B 16-bit layout half offset

  v8f acc[4];
#pragma unroll
  for (int nt = 0; nt < 4; ++nt)
#pragma unroll
    for (int r = 0; r < 8; ++r) acc[nt][r] = 0.0f;

#if defined(USE_TDM)
  // Prologue: DMA slab 0 into buffer 0 while nothing else is in flight.
  if (wave == 0) {
    tdm_stage_b(bt, (unsigned)(uintptr_t)&smem_raw[0]);
  }
#endif

  for (int p = 0; p < NPHASE; ++p) {
    const int k0 = p * KPHASE;
    bf16_t* sB = (bf16_t*)&smem_raw[(p & 1) * STAGE_BYTES];

#if defined(USE_TDM)
    if (wave == 0) {
      if (p + 1 < NPHASE) {
        // Prefetch next slab into the other buffer (its last readers finished
        // at the end-of-phase barrier of iteration p-1), then require slab p
        // to have landed: <=1 outstanding tensor op.
        tdm_stage_b(bt + (size_t)(p + 1) * KPHASE,
                    (unsigned)(uintptr_t)&smem_raw[((p + 1) & 1) * STAGE_BYTES]);
        __builtin_amdgcn_s_wait_tensorcnt(1);
      } else {
        __builtin_amdgcn_s_wait_tensorcnt(0);
      }
    }
    __syncthreads();   // slab p visible to all waves
#else
    // Cooperative staging fallback: 2 planes * 64 rows * 256B in 16B quanta.
    __syncthreads();   // previous readers of this buffer done
    for (int q = tid; q < 2 * E_EXP * (KPHASE / 8); q += WAVES_PB * 32) {
      int plane = q >> 10;
      int n     = (q >> 4) & (E_EXP - 1);
      int kq    = q & 15;
      const uint4* src = (const uint4*)(bt + ((size_t)(plane * E_EXP + n)) * D_DIM +
                                        k0 + kq * 8);
      *(uint4*)(sB + (plane * E_EXP + n) * LDS_ROW + kq * 8) = *src;
    }
    __syncthreads();
#endif

#pragma unroll
    for (int ks = 0; ks < KPHASE; ks += KCHUNK) {
      const int kk = k0 + ks;
      // ---- A fragment: 16 f32 per lane -> split into hi/lo bf16 -------------
      float4 f0 = *(const float4*)(xrow + kk + kloA);
      float4 f1 = *(const float4*)(xrow + kk + kloA + 4);
      float4 f2 = *(const float4*)(xrow + kk + kloA + 16);
      float4 f3 = *(const float4*)(xrow + kk + kloA + 20);
      float fa[16] = {f0.x, f0.y, f0.z, f0.w, f1.x, f1.y, f1.z, f1.w,
                      f2.x, f2.y, f2.z, f2.w, f3.x, f3.y, f3.z, f3.w};
      v16bf a_hi, a_lo;
#pragma unroll
      for (int j = 0; j < 16; ++j) {
        __bf16 h = (__bf16)fa[j];
        a_hi[j] = h;
        a_lo[j] = (__bf16)(fa[j] - (float)h);
      }
      // ---- B fragments for all 4 N-tiles (hi + lo planes) -------------------
      v16bf bHi[4], bLo[4];
#pragma unroll
      for (int nt = 0; nt < 4; ++nt) {
        const bf16_t* bh = sB + (nt * 16 + l15) * LDS_ROW + ks + kloB;
        const bf16_t* bl = bh + E_EXP * LDS_ROW;   // lo plane
        v8bf h0 = *(const v8bf*)(bh);
        v8bf h1 = *(const v8bf*)(bh + 16);
        v8bf l0 = *(const v8bf*)(bl);
        v8bf l1 = *(const v8bf*)(bl + 16);
        bHi[nt] = __builtin_shufflevector(h0, h1, 0, 1, 2, 3, 4, 5, 6, 7,
                                          8, 9, 10, 11, 12, 13, 14, 15);
        bLo[nt] = __builtin_shufflevector(l0, l1, 0, 1, 2, 3, 4, 5, 6, 7,
                                          8, 9, 10, 11, 12, 13, 14, 15);
      }
      // ---- bf16x3 compensated MACs, term-major so dependent WMMAs on the same
      //      accumulator are 4 instructions apart (hides XDL RAW latency) ------
#pragma unroll
      for (int nt = 0; nt < 4; ++nt)
        acc[nt] = __builtin_amdgcn_wmma_f32_16x16x32_bf16(
            false, a_hi, false, bHi[nt], (short)0, acc[nt], false, false);
#pragma unroll
      for (int nt = 0; nt < 4; ++nt)
        acc[nt] = __builtin_amdgcn_wmma_f32_16x16x32_bf16(
            false, a_hi, false, bLo[nt], (short)0, acc[nt], false, false);
#pragma unroll
      for (int nt = 0; nt < 4; ++nt)
        acc[nt] = __builtin_amdgcn_wmma_f32_16x16x32_bf16(
            false, a_lo, false, bHi[nt], (short)0, acc[nt], false, false);
    }
    __syncthreads();   // all waves done with slab p before its buffer is reused
  }

  // ---- Epilogue: scatter C tiles to LDS (layout: VGPR r -> M=r / M=r+8) ------
  const int mbase = wave * TOK_PW + h16 * 8;
#pragma unroll
  for (int nt = 0; nt < 4; ++nt)
#pragma unroll
    for (int r = 0; r < 8; ++r)
      sL[(mbase + r) * 68 + nt * 16 + l15] = acc[nt][r];
  __syncthreads();

  if (tid < TOK_PB) {
    topk8_softmax_store(sL + tid * 68, blockIdx.x * TOK_PB + tid, out_w, out_i);
  }
}

// ---------------------------------------------------------------------------
// Fallback (no workspace): plain f32 dot per (token, expert). Slow but exact.
// ---------------------------------------------------------------------------
__global__ __launch_bounds__(E_EXP) void router_fallback(
    const float* __restrict__ x, const float* __restrict__ wde,
    float* __restrict__ out_w, int* __restrict__ out_i) {
  __shared__ float sl[E_EXP];
  const int token = blockIdx.x;
  const int e = threadIdx.x;
  const float* xr = x + (size_t)token * D_DIM;
  float acc = 0.0f;
  for (int d = 0; d < D_DIM; ++d) acc = __fmaf_rn(xr[d], wde[(size_t)d * E_EXP + e], acc);
  sl[e] = acc;
  __syncthreads();
  if (e == 0) topk8_softmax_store(sl, token, out_w, out_i);
}

// ---------------------------------------------------------------------------
extern "C" void kernel_launch(void* const* d_in, const int* in_sizes, int n_in,
                              void* d_out, int out_size, void* d_ws, size_t ws_size,
                              hipStream_t stream) {
  (void)in_sizes; (void)n_in; (void)out_size;
  const float* x   = (const float*)d_in[0];    // [8192, 4096] f32
  const float* wde = (const float*)d_in[1];    // [4096, 64]   f32
  float* out_w = (float*)d_out;                             // [8192, 8] f32
  int*   out_i = (int*)((float*)d_out + (size_t)T_TOK * TOPK_K);  // [8192, 8] i32

  const size_t ws_needed = (size_t)2 * E_EXP * D_DIM * sizeof(bf16_t);  // 1 MB
  if (ws_size >= ws_needed) {
    bf16_t* bt = (bf16_t*)d_ws;   // [2][E][D] hi/lo planes, transposed
    router_prep_b<<<(E_EXP * D_DIM + 255) / 256, 256, 0, stream>>>(wde, bt);
    router_gemm_topk<<<T_TOK / TOK_PB, WAVES_PB * 32, 0, stream>>>(x, bt, out_w, out_i);
  } else {
    router_fallback<<<T_TOK, E_EXP, 0, stream>>>(x, wde, out_w, out_i);
  }
}
